// TransMIL_11244224381150
// MI455X (gfx1250) — compile-verified, matrix operand order checked by reading, not run on hardware
//
#include <hip/hip_runtime.h>
#include <hip/hip_bf16.h>
#include <cstdint>
#include <cstddef>

typedef __attribute__((ext_vector_type(16))) _Float16 v16h;
typedef __attribute__((ext_vector_type(8)))  float    v8f;
typedef __attribute__((ext_vector_type(2)))  float    v2f;

#ifndef __has_builtin
#define __has_builtin(x) 0
#endif

// Async global->LDS (CDNA5 ASYNCcnt path), guarded so compile never regresses.
// Probe round 2 revealed the builtin's param type: int4 (vector_size 16) pointers,
// global (AS1) source, LDS (AS3) destination.
#if __has_builtin(__builtin_amdgcn_global_load_async_to_lds_b128)
#define USE_ASYNC_LDS 1
typedef int v4i_vs __attribute__((__vector_size__(4 * sizeof(int))));
typedef __attribute__((address_space(1))) v4i_vs* gas_v4i_p;
typedef __attribute__((address_space(3))) v4i_vs* las_v4i_p;
#define ASYNC_LOAD_B128(gp, lp) \
  __builtin_amdgcn_global_load_async_to_lds_b128((gas_v4i_p)(gp), (las_v4i_p)(lp), 0, 0)
#if __has_builtin(__builtin_amdgcn_s_wait_asynccnt)
#define WAIT_ASYNC() __builtin_amdgcn_s_wait_asynccnt(0)
#else
#define WAIT_ASYNC() asm volatile("s_wait_asynccnt 0" ::: "memory")
#endif
#else
#define USE_ASYNC_LDS 0
#endif

// ---------------------------------------------------------------- convert
__global__ void k_f32_to_f16(const float* __restrict__ src, _Float16* __restrict__ dst, long n) {
  long i = (long)blockIdx.x * blockDim.x + threadIdx.x;
  if (i < n) dst[i] = (_Float16)src[i];
}

// ---------------------------------------------------------------- f16 WMMA GEMM
// C(M,N) = A(M,K) * B(K,N) [+bias] [relu], A/B f16 row-major, accum f32.
// Block: 256 thr = 8 waves, tile 64x64, K-step 32 (v_wmma_f32_16x16x32_f16).
// Double-buffered LDS; async global->LDS copies overlap the next tile with WMMA.
template<int RELU, int OUTF16>
__global__ __launch_bounds__(256) void k_gemm_f16(
    const _Float16* __restrict__ A, const _Float16* __restrict__ Bm,
    const float* __restrict__ bias, void* __restrict__ Cvoid,
    int M, int N, int K)
{
  __shared__ __align__(16) _Float16 As[2][64 * 32];
  __shared__ __align__(16) _Float16 Bs[2][32 * 64];
  const int tid  = threadIdx.x;
  const int lane = tid & 31, wave = tid >> 5;
  const int m0 = blockIdx.y * 64, n0 = blockIdx.x * 64;
  const int sc = wave & 3, sr0 = (wave >> 2) * 2;
  v8f acc[2];
  #pragma unroll
  for (int i = 0; i < 8; ++i) { acc[0][i] = 0.f; acc[1][i] = 0.f; }
  const int ar = tid >> 2, ac = (tid & 3) * 8;
  const int br = tid >> 3, bc = (tid & 7) * 8;
  const _Float16* Ag = A  + (size_t)(m0 + ar) * K + ac;   // + k0
  const _Float16* Bg = Bm + (size_t)br * N + n0 + bc;     // + k0*N

#if USE_ASYNC_LDS
  ASYNC_LOAD_B128(Ag, &As[0][ar * 32 + ac]);
  ASYNC_LOAD_B128(Bg, &Bs[0][br * 64 + bc]);
#endif

  int buf = 0;
  for (int k0 = 0; k0 < K; k0 += 32, buf ^= 1) {
#if USE_ASYNC_LDS
    WAIT_ASYNC();
    __syncthreads();          // publishes tile k0; frees buffer buf^1 for refill
    if (k0 + 32 < K) {
      ASYNC_LOAD_B128(Ag + (k0 + 32),             &As[buf ^ 1][ar * 32 + ac]);
      ASYNC_LOAD_B128(Bg + (size_t)(k0 + 32) * N, &Bs[buf ^ 1][br * 64 + bc]);
    }
#else
    *(uint4*)(&As[buf][ar * 32 + ac]) = *(const uint4*)(Ag + k0);
    *(uint4*)(&Bs[buf][br * 64 + bc]) = *(const uint4*)(Bg + (size_t)k0 * N);
    if (k0 + 32 < K) {
      __builtin_prefetch(Ag + k0 + 32);
      __builtin_prefetch(Bg + (size_t)(k0 + 32) * N);
    }
    __syncthreads();
#endif
    const _Float16* Al = As[buf];
    const _Float16* Bl = Bs[buf];
    const int kb = (lane < 16) ? 0 : 8;   // ISA 7.12.2 16-bit A/B layout
    const int r0 = lane & 15;
    v16h bf;
    {
      const int col = sc * 16 + r0;
      #pragma unroll
      for (int v = 0; v < 4; ++v) {
        bf[2 * v]         = Bl[(kb + 2 * v) * 64 + col];
        bf[2 * v + 1]     = Bl[(kb + 2 * v + 1) * 64 + col];
        bf[8 + 2 * v]     = Bl[(16 + kb + 2 * v) * 64 + col];
        bf[8 + 2 * v + 1] = Bl[(16 + kb + 2 * v + 1) * 64 + col];
      }
    }
    #pragma unroll
    for (int s = 0; s < 2; ++s) {
      const int row = (sr0 + s) * 16 + r0;
      v16h af;
      #pragma unroll
      for (int v = 0; v < 4; ++v) {
        af[2 * v]         = Al[row * 32 + kb + 2 * v];
        af[2 * v + 1]     = Al[row * 32 + kb + 2 * v + 1];
        af[8 + 2 * v]     = Al[row * 32 + 16 + kb + 2 * v];
        af[8 + 2 * v + 1] = Al[row * 32 + 16 + kb + 2 * v + 1];
      }
      acc[s] = __builtin_amdgcn_wmma_f32_16x16x32_f16(false, af, false, bf,
                                                      (short)0, acc[s], false, false);
    }
#if !USE_ASYNC_LDS
    __syncthreads();
#endif
  }
  const int col = n0 + sc * 16 + (lane & 15);
  const float bv = bias ? bias[col] : 0.f;
  float*    Cf = (float*)Cvoid;
  _Float16* Ch = (_Float16*)Cvoid;
  #pragma unroll
  for (int s = 0; s < 2; ++s)
    #pragma unroll
    for (int r = 0; r < 8; ++r) {
      const int row = m0 + (sr0 + s) * 16 + ((lane < 16) ? r : 8 + r);
      float val = acc[s][r] + bv;
      if (RELU) val = fmaxf(val, 0.f);
      if (OUTF16) Ch[(size_t)row * N + col] = (_Float16)val;
      else        Cf[(size_t)row * N + col] = val;
    }
}

// ---------------------------------------------------------------- fp32 WMMA batched matmul
// C = c_diag*I + c_mul*(A@B), batched over blockIdx.z (v_wmma_f32_16x16x4_f32).
__global__ __launch_bounds__(256) void k_bmm_f32(
    const float* __restrict__ A, const float* __restrict__ Bm, float* __restrict__ C,
    int M, int N, int K, float c_diag, float c_mul)
{
  __shared__ __align__(16) float As[64 * 16];
  __shared__ __align__(16) float Bs[16 * 64];
  const int tid = threadIdx.x, lane = tid & 31, wave = tid >> 5;
  const int m0 = blockIdx.y * 64, n0 = blockIdx.x * 64;
  const float* Ab = A  + (size_t)blockIdx.z * M * K;
  const float* Bb = Bm + (size_t)blockIdx.z * K * N;
  float*       Cb = C  + (size_t)blockIdx.z * M * N;
  const int sc = wave & 3, sr0 = (wave >> 2) * 2;
  v8f acc[2];
  #pragma unroll
  for (int i = 0; i < 8; ++i) { acc[0][i] = 0.f; acc[1][i] = 0.f; }
  const int ar = tid >> 2,  ac = (tid & 3) * 4;
  const int br = tid >> 4,  bc = (tid & 15) * 4;
  for (int k0 = 0; k0 < K; k0 += 16) {
    *(float4*)(As + ar * 16 + ac) = *(const float4*)(Ab + (size_t)(m0 + ar) * K + k0 + ac);
    *(float4*)(Bs + br * 64 + bc) = *(const float4*)(Bb + (size_t)(k0 + br) * N + n0 + bc);
    __syncthreads();
    const int r0 = lane & 15;
    const int kh = (lane < 16) ? 0 : 2;   // ISA 7.12.2 32-bit A 16x4 layout
    #pragma unroll
    for (int kk = 0; kk < 4; ++kk) {
      v2f bf;
      const int col = sc * 16 + r0;
      bf[0] = Bs[(kk * 4 + kh) * 64 + col];
      bf[1] = Bs[(kk * 4 + kh + 1) * 64 + col];
      #pragma unroll
      for (int s = 0; s < 2; ++s) {
        const int row = (sr0 + s) * 16 + r0;
        v2f af;
        af[0] = As[row * 16 + kk * 4 + kh];
        af[1] = As[row * 16 + kk * 4 + kh + 1];
        acc[s] = __builtin_amdgcn_wmma_f32_16x16x4_f32(false, af, false, bf,
                                                       (short)0, acc[s], false, false);
      }
    }
    __syncthreads();
  }
  const int colg = n0 + sc * 16 + (lane & 15);
  #pragma unroll
  for (int s = 0; s < 2; ++s)
    #pragma unroll
    for (int r = 0; r < 8; ++r) {
      const int rowg = m0 + (sr0 + s) * 16 + ((lane < 16) ? r : 8 + r);
      Cb[(size_t)rowg * N + colg] = c_mul * acc[s][r] + ((rowg == colg) ? c_diag : 0.f);
    }
}

// ---------------------------------------------------------------- token assembly
__global__ void k_build_h(const float* __restrict__ h0, const float* __restrict__ cls,
                          float* __restrict__ h) {
  long idx = (long)blockIdx.x * 256 + threadIdx.x;
  if (idx >= 2L * 16130 * 512) return;
  int c = (int)(idx & 511);
  long row = idx >> 9;
  int b = (int)(row / 16130), p = (int)(row % 16130);
  float v;
  if (p == 0) v = cls[c];
  else { int i = p - 1; if (i >= 16000) i -= 16000; v = h0[((size_t)b * 16000 + i) * 512 + c]; }
  h[idx] = v;
}

// ---------------------------------------------------------------- layernorm -> f16 (padded to 16384 rows, pad at FRONT)
__global__ __launch_bounds__(256) void k_layernorm(
    const float* __restrict__ h, const float* __restrict__ g, const float* __restrict__ bt,
    _Float16* __restrict__ xln)
{
  __shared__ float red[256];
  const int row = blockIdx.x;                 // 0..32259
  const int b = row / 16130, p = row % 16130;
  const float* x = h + (size_t)row * 512;
  const int tid = threadIdx.x;
  float a0 = x[tid], a1 = x[tid + 256];
  red[tid] = a0 + a1; __syncthreads();
  for (int s = 128; s > 0; s >>= 1) { if (tid < s) red[tid] += red[tid + s]; __syncthreads(); }
  float mu = red[0] * (1.f / 512.f); __syncthreads();
  float d0 = a0 - mu, d1 = a1 - mu;
  red[tid] = d0 * d0 + d1 * d1; __syncthreads();
  for (int s = 128; s > 0; s >>= 1) { if (tid < s) red[tid] += red[tid + s]; __syncthreads(); }
  float rstd = rsqrtf(red[0] * (1.f / 512.f) + 1e-5f);
  _Float16* dst = xln + ((size_t)b * 16384 + 254 + p) * 512;
  dst[tid]       = (_Float16)(d0 * rstd * g[tid] + bt[tid]);
  dst[tid + 256] = (_Float16)(d1 * rstd * g[tid + 256] + bt[tid + 256]);
}

__global__ void k_zero_pad(_Float16* __restrict__ xln) {
  int idx = blockIdx.x * 256 + threadIdx.x;   // 2*254*512
  if (idx >= 2 * 254 * 512) return;
  int b = idx / (254 * 512), r = idx % (254 * 512);
  xln[(size_t)b * 16384 * 512 + r] = (_Float16)0.f;
}

// ---------------------------------------------------------------- landmarks (mean of 64 consecutive tokens)
__global__ void k_landmarks(const _Float16* __restrict__ qkv,
                            _Float16* __restrict__ ql, _Float16* __restrict__ kl) {
  const int i = blockIdx.x, bh = blockIdx.y, d = threadIdx.x;  // block=64
  const int b = bh >> 3, hh = bh & 7;
  float sq = 0.f, sk = 0.f;
  for (int j = 0; j < 64; ++j) {
    size_t base = ((size_t)(b * 16384) + i * 64 + j) * 1536 + hh * 64 + d;
    sq += (float)qkv[base];
    sk += (float)qkv[base + 512];
  }
  ql[((size_t)bh * 256 + i) * 64 + d] = (_Float16)(sq * (1.f / 64.f));
  kl[((size_t)bh * 256 + i) * 64 + d] = (_Float16)(sk * (1.f / 64.f));
}

// ---------------------------------------------------------------- attn2 = softmax(q_l k_l^T / 8)
__global__ __launch_bounds__(256) void k_attn2(const _Float16* __restrict__ ql,
                                               const _Float16* __restrict__ kl,
                                               float* __restrict__ X) {
  __shared__ float qs[64];
  __shared__ float red[256];
  const int i = blockIdx.x, bh = blockIdx.y, tid = threadIdx.x;
  if (tid < 64) qs[tid] = 0.125f * (float)ql[((size_t)bh * 256 + i) * 64 + tid];
  __syncthreads();
  const _Float16* kr = kl + ((size_t)bh * 256 + tid) * 64;
  float s = 0.f;
  #pragma unroll 16
  for (int d = 0; d < 64; ++d) s += qs[d] * (float)kr[d];
  red[tid] = s; __syncthreads();
  for (int w = 128; w > 0; w >>= 1) { if (tid < w) red[tid] = fmaxf(red[tid], red[tid + w]); __syncthreads(); }
  float m = red[0]; __syncthreads();
  float p = __expf(s - m);
  red[tid] = p; __syncthreads();
  for (int w = 128; w > 0; w >>= 1) { if (tid < w) red[tid] += red[tid + w]; __syncthreads(); }
  X[((size_t)bh * 256 + i) * 256 + tid] = p / red[0];
}

// ---------------------------------------------------------------- pinv init: rows of softmax sum to 1 -> scale = max col-sum
__global__ void k_init_scalar(float* p) { if (threadIdx.x == 0 && blockIdx.x == 0) *p = 0.f; }

__global__ __launch_bounds__(256) void k_colsum_max(const float* __restrict__ X, float* __restrict__ cmax) {
  const int bh = blockIdx.x, j = threadIdx.x;
  float s = 0.f;
  for (int i = 0; i < 256; ++i) s += X[((size_t)bh * 256 + i) * 256 + j];
  atomicMax((unsigned int*)cmax, __float_as_uint(s));   // all values positive
}

__global__ __launch_bounds__(256) void k_zinit(const float* __restrict__ X, const float* __restrict__ cmax,
                                               float* __restrict__ Z) {
  const int i = blockIdx.x, bh = blockIdx.y, j = threadIdx.x;
  Z[((size_t)bh * 256 + i) * 256 + j] = X[((size_t)bh * 256 + j) * 256 + i] / (*cmax);
}

// ---------------------------------------------------------------- out3 = softmax(q_l K^T / 8) @ V  (flash 2-pass)
__global__ __launch_bounds__(256) void k_attn3v(
    const _Float16* __restrict__ qkv, const _Float16* __restrict__ ql, float* __restrict__ out3)
{
  __shared__ float qs[64];
  __shared__ float red[256];
  __shared__ float ps[256];
  const int i = blockIdx.x, bh = blockIdx.y, tid = threadIdx.x;
  const int b = bh >> 3, hh = bh & 7;
  if (tid < 64) qs[tid] = 0.125f * (float)ql[((size_t)bh * 256 + i) * 64 + tid];
  __syncthreads();
  float lmax = -3.0e38f;
  for (int j = tid; j < 16384; j += 256) {
    const _Float16* kr = qkv + ((size_t)(b * 16384) + j) * 1536 + 512 + hh * 64;
    float s = 0.f;
    #pragma unroll 16
    for (int d = 0; d < 64; ++d) s += qs[d] * (float)kr[d];
    lmax = fmaxf(lmax, s);
  }
  red[tid] = lmax; __syncthreads();
  for (int w = 128; w > 0; w >>= 1) { if (tid < w) red[tid] = fmaxf(red[tid], red[tid + w]); __syncthreads(); }
  const float m = red[0]; __syncthreads();
  float lsum = 0.f, acc = 0.f;
  for (int c0 = 0; c0 < 16384; c0 += 256) {
    const int j = c0 + tid;
    const _Float16* kr = qkv + ((size_t)(b * 16384) + j) * 1536 + 512 + hh * 64;
    float s = 0.f;
    #pragma unroll 16
    for (int d = 0; d < 64; ++d) s += qs[d] * (float)kr[d];
    const float p = __expf(s - m);
    ps[tid] = p; lsum += p;
    __syncthreads();
    if (tid < 64) {
      for (int jj = 0; jj < 256; ++jj)
        acc += ps[jj] * (float)qkv[((size_t)(b * 16384) + c0 + jj) * 1536 + 1024 + hh * 64 + tid];
    }
    __syncthreads();
  }
  red[tid] = lsum; __syncthreads();
  for (int w = 128; w > 0; w >>= 1) { if (tid < w) red[tid] += red[tid + w]; __syncthreads(); }
  if (tid < 64) out3[((size_t)bh * 256 + i) * 64 + tid] = acc / red[0];
}

// ---------------------------------------------------------------- per-token: softmax(q k_l^T/8) @ agg + 33-tap dwconv(v)
__global__ __launch_bounds__(256) void k_attn1_out(
    const _Float16* __restrict__ qkv, const _Float16* __restrict__ kl,
    const float* __restrict__ agg, const float* __restrict__ resw,
    _Float16* __restrict__ xout)
{
  __shared__ _Float16 aggs[256 * 64];
  __shared__ float ps[8][256];
  __shared__ float qsh[8][64];
  __shared__ float wc[33];
  const int tid = threadIdx.x, lane = tid & 31, wave = tid >> 5;
  const int g = blockIdx.x, bh = blockIdx.y;
  const int b = bh >> 3, hh = bh & 7;
  for (int idx = tid; idx < 256 * 64; idx += 256) aggs[idx] = (_Float16)agg[(size_t)bh * 16384 + idx];
  if (tid < 33) wc[tid] = resw[hh * 33 + tid];
  __syncthreads();
  for (int tw = 0; tw < 16; ++tw) {
    const int t = g * 128 + tw * 8 + wave;
    const size_t qbase = ((size_t)(b * 16384) + t) * 1536 + hh * 64;
    qsh[wave][lane]      = 0.125f * (float)qkv[qbase + lane];
    qsh[wave][lane + 32] = 0.125f * (float)qkv[qbase + lane + 32];
    float lg[8];
    float lmax = -3.0e38f;
    #pragma unroll
    for (int r = 0; r < 8; ++r) {
      const int i = lane + 32 * r;
      const _Float16* kr = kl + ((size_t)bh * 256 + i) * 64;
      float s = 0.f;
      #pragma unroll 16
      for (int d = 0; d < 64; ++d) s += qsh[wave][d] * (float)kr[d];
      lg[r] = s; lmax = fmaxf(lmax, s);
    }
    #pragma unroll
    for (int off = 16; off > 0; off >>= 1) lmax = fmaxf(lmax, __shfl_xor(lmax, off, 32));
    float lsum = 0.f;
    #pragma unroll
    for (int r = 0; r < 8; ++r) { float p = __expf(lg[r] - lmax); ps[wave][lane + 32 * r] = p; lsum += p; }
    #pragma unroll
    for (int off = 16; off > 0; off >>= 1) lsum += __shfl_xor(lsum, off, 32);
    const float inv = 1.f / lsum;
    #pragma unroll
    for (int e = 0; e < 2; ++e) {
      const int d = lane + 32 * e;
      float y = 0.f;
      #pragma unroll 8
      for (int i = 0; i < 256; ++i) y += ps[wave][i] * (float)aggs[i * 64 + d];
      float rs = 0.f;
      #pragma unroll
      for (int r = 0; r < 33; ++r) {
        const int tt = t + r - 16;
        if (tt >= 0 && tt < 16384)
          rs += wc[r] * (float)qkv[((size_t)(b * 16384) + tt) * 1536 + 1024 + hh * 64 + d];
      }
      xout[((size_t)(b * 16384) + t) * 512 + hh * 64 + d] = (_Float16)(y * inv + rs);
    }
  }
}

// ---------------------------------------------------------------- residual: h += yatt[last 16130 rows]
__global__ void k_residual(float* __restrict__ h, const float* __restrict__ yatt) {
  long idx = (long)blockIdx.x * 256 + threadIdx.x;
  if (idx >= 2L * 16130 * 512) return;
  int c = (int)(idx & 511);
  long row = idx >> 9;
  int b = (int)(row / 16130), p = (int)(row % 16130);
  h[idx] += yatt[((size_t)b * 16384 + 254 + p) * 512 + c];
}

// ---------------------------------------------------------------- PPEG depthwise 7/5/3 + identity
__global__ __launch_bounds__(128) void k_ppeg(
    const float* __restrict__ h, float* __restrict__ h2,
    const float* __restrict__ w7, const float* __restrict__ b7,
    const float* __restrict__ w5, const float* __restrict__ b5,
    const float* __restrict__ w3, const float* __restrict__ b3)
{
  __shared__ float rows[7][136];
  __shared__ float w7s[49], w5s[25], w3s[9];
  const int r = blockIdx.x;                  // 0..126
  const int b = blockIdx.y >> 9, c = blockIdx.y & 511;
  const int tid = threadIdx.x;
  if (tid < 49) w7s[tid] = w7[c * 49 + tid];
  if (tid < 25) w5s[tid] = w5[c * 25 + tid];
  if (tid < 9)  w3s[tid] = w3[c * 9 + tid];
  for (int dr = 0; dr < 7; ++dr) {
    const int rr = r + dr - 3;
    for (int col = tid; col < 133; col += 128) {
      const int cc = col - 3;
      float v = 0.f;
      if (rr >= 0 && rr < 127 && cc >= 0 && cc < 127)
        v = h[((size_t)b * 16130 + 1 + rr * 127 + cc) * 512 + c];
      rows[dr][col] = v;
    }
  }
  __syncthreads();
  if (tid < 127) {
    const int q = tid;
    float acc = rows[3][q + 3];
    float a7 = 0.f, a5 = 0.f, a3 = 0.f;
    #pragma unroll
    for (int dr = 0; dr < 7; ++dr)
      #pragma unroll
      for (int dc = 0; dc < 7; ++dc) a7 += w7s[dr * 7 + dc] * rows[dr][q + dc];
    #pragma unroll
    for (int dr = 0; dr < 5; ++dr)
      #pragma unroll
      for (int dc = 0; dc < 5; ++dc) a5 += w5s[dr * 5 + dc] * rows[dr + 1][q + 1 + dc];
    #pragma unroll
    for (int dr = 0; dr < 3; ++dr)
      #pragma unroll
      for (int dc = 0; dc < 3; ++dc) a3 += w3s[dr * 3 + dc] * rows[dr + 2][q + 2 + dc];
    acc += a7 + b7[c] + a5 + b5[c] + a3 + b3[c];
    h2[((size_t)b * 16130 + 1 + r * 127 + q) * 512 + c] = acc;
  }
}

__global__ void k_copy_cls(const float* __restrict__ h, float* __restrict__ h2) {
  int idx = blockIdx.x * 256 + threadIdx.x;   // 1024
  if (idx >= 1024) return;
  int b = idx >> 9, c = idx & 511;
  h2[(size_t)b * 16130 * 512 + c] = h[(size_t)b * 16130 * 512 + c];
}

// ---------------------------------------------------------------- final LN(cls token) @ fc2
__global__ __launch_bounds__(256) void k_final(
    const float* __restrict__ h, const float* __restrict__ g, const float* __restrict__ bt,
    const float* __restrict__ w, const float* __restrict__ fb, float* __restrict__ out)
{
  __shared__ float red[256];
  const int b = blockIdx.x, tid = threadIdx.x;
  const float* x = h + (size_t)b * 16130 * 512;
  float a0 = x[tid], a1 = x[tid + 256];
  red[tid] = a0 + a1; __syncthreads();
  for (int s = 128; s > 0; s >>= 1) { if (tid < s) red[tid] += red[tid + s]; __syncthreads(); }
  float mu = red[0] * (1.f / 512.f); __syncthreads();
  float d0 = a0 - mu, d1 = a1 - mu;
  red[tid] = d0 * d0 + d1 * d1; __syncthreads();
  for (int s = 128; s > 0; s >>= 1) { if (tid < s) red[tid] += red[tid + s]; __syncthreads(); }
  float rstd = rsqrtf(red[0] * (1.f / 512.f) + 1e-5f); __syncthreads();
  float dot = (d0 * rstd * g[tid] + bt[tid]) * w[tid]
            + (d1 * rstd * g[tid + 256] + bt[tid + 256]) * w[tid + 256];
  red[tid] = dot; __syncthreads();
  for (int s = 128; s > 0; s >>= 1) { if (tid < s) red[tid] += red[tid + s]; __syncthreads(); }
  if (tid == 0) out[b] = red[0] + fb[0];
}

// ================================================================ host
extern "C" void kernel_launch(void* const* d_in, const int* in_sizes, int n_in,
                              void* d_out, int out_size, void* d_ws, size_t ws_size,
                              hipStream_t stream)
{
  (void)in_sizes; (void)n_in; (void)out_size; (void)ws_size;
  const float* features = (const float*)d_in[0];
  const float* fc1_w = (const float*)d_in[1];
  const float* fc1_b = (const float*)d_in[2];
  const float* cls   = (const float*)d_in[3];
  const float* l1_g  = (const float*)d_in[4];
  const float* l1_bt = (const float*)d_in[5];
  const float* l1_qkv_w = (const float*)d_in[6];
  const float* l1_out_w = (const float*)d_in[7];
  const float* l1_out_b = (const float*)d_in[8];
  const float* l1_res_w = (const float*)d_in[9];
  const float* w7 = (const float*)d_in[10]; const float* b7 = (const float*)d_in[11];
  const float* w5 = (const float*)d_in[12]; const float* b5 = (const float*)d_in[13];
  const float* w3 = (const float*)d_in[14]; const float* b3 = (const float*)d_in[15];
  const float* l2_g  = (const float*)d_in[16]; const float* l2_bt = (const float*)d_in[17];
  const float* l2_qkv_w = (const float*)d_in[18];
  const float* l2_out_w = (const float*)d_in[19];
  const float* l2_out_b = (const float*)d_in[20];
  const float* l2_res_w = (const float*)d_in[21];
  const float* n_g  = (const float*)d_in[22]; const float* n_bt = (const float*)d_in[23];
  const float* fc2_w = (const float*)d_in[24]; const float* fc2_b = (const float*)d_in[25];

  char* ws = (char*)d_ws;
  size_t off = 0;
  auto alloc = [&](size_t bytes) -> void* {
    void* p = ws + off;
    off += (bytes + 255) & ~(size_t)255;
    return p;
  };
  _Float16* feat16  = (_Float16*)alloc(32000UL * 1024 * 2);
  _Float16* fc1w16  = (_Float16*)alloc(1024UL * 512 * 2);
  _Float16* qkvw16a = (_Float16*)alloc(512UL * 1536 * 2);
  _Float16* qkvw16b = (_Float16*)alloc(512UL * 1536 * 2);
  _Float16* outw16a = (_Float16*)alloc(512UL * 512 * 2);
  _Float16* outw16b = (_Float16*)alloc(512UL * 512 * 2);
  float*    h0      = (float*)alloc(32000UL * 512 * 4);
  float*    h       = (float*)alloc(2UL * 16130 * 512 * 4);
  float*    h2      = (float*)alloc(2UL * 16130 * 512 * 4);
  _Float16* xln     = (_Float16*)alloc(2UL * 16384 * 512 * 2);
  _Float16* qkv16   = (_Float16*)alloc(2UL * 16384 * 1536 * 2);
  _Float16* ql16    = (_Float16*)alloc(16UL * 256 * 64 * 2);
  _Float16* kl16    = (_Float16*)alloc(16UL * 256 * 64 * 2);
  float*    X       = (float*)alloc(16UL * 256 * 256 * 4);
  float*    Za      = (float*)alloc(16UL * 256 * 256 * 4);
  float*    Zb      = (float*)alloc(16UL * 256 * 256 * 4);
  float*    XZ      = (float*)alloc(16UL * 256 * 256 * 4);
  float*    T1      = (float*)alloc(16UL * 256 * 256 * 4);
  float*    T2      = (float*)alloc(16UL * 256 * 256 * 4);
  float*    out3    = (float*)alloc(16UL * 256 * 64 * 4);
  float*    agg     = (float*)alloc(16UL * 256 * 64 * 4);
  _Float16* xout16  = (_Float16*)alloc(2UL * 16384 * 512 * 2);
  float*    yatt    = (float*)alloc(2UL * 16384 * 512 * 4);
  float*    cmax    = (float*)alloc(256);

  auto cvt = [&](const float* s, _Float16* d, long n) {
    k_f32_to_f16<<<dim3((unsigned)((n + 255) / 256)), dim3(256), 0, stream>>>(s, d, n);
  };
  cvt(features, feat16, 32000L * 1024);
  cvt(fc1_w, fc1w16, 1024L * 512);
  cvt(l1_qkv_w, qkvw16a, 512L * 1536);
  cvt(l2_qkv_w, qkvw16b, 512L * 1536);
  cvt(l1_out_w, outw16a, 512L * 512);
  cvt(l2_out_w, outw16b, 512L * 512);

  // fc1 + relu : (32000,1024)@(1024,512)
  k_gemm_f16<1, 0><<<dim3(8, 500), 256, 0, stream>>>(feat16, fc1w16, fc1_b, h0, 32000, 512, 1024);
  k_build_h<<<dim3((2 * 16130 * 512 + 255) / 256), 256, 0, stream>>>(h0, cls, h);

  auto layer = [&](float* hbuf, const float* g, const float* bt, _Float16* qkvw,
                   _Float16* outw, const float* outb, const float* resw) {
    k_zero_pad<<<dim3((2 * 254 * 512 + 255) / 256), 256, 0, stream>>>(xln);
    k_layernorm<<<dim3(2 * 16130), 256, 0, stream>>>(hbuf, g, bt, xln);
    k_gemm_f16<0, 1><<<dim3(24, 512), 256, 0, stream>>>(xln, qkvw, nullptr, qkv16, 32768, 1536, 512);
    k_landmarks<<<dim3(256, 16), 64, 0, stream>>>(qkv16, ql16, kl16);
    k_attn2<<<dim3(256, 16), 256, 0, stream>>>(ql16, kl16, X);
    k_init_scalar<<<1, 1, 0, stream>>>(cmax);
    k_colsum_max<<<dim3(16), 256, 0, stream>>>(X, cmax);
    k_zinit<<<dim3(256, 16), 256, 0, stream>>>(X, cmax, Za);
    float* Zc = Za; float* Zo = Zb;
    for (int it = 0; it < 6; ++it) {    // Newton-Schulz, fp32 WMMA
      k_bmm_f32<<<dim3(4, 4, 16), 256, 0, stream>>>(X,  Zc, XZ, 256, 256, 256, 0.f,  1.f);
      k_bmm_f32<<<dim3(4, 4, 16), 256, 0, stream>>>(X,  Zc, T1, 256, 256, 256, 7.f, -1.f);
      k_bmm_f32<<<dim3(4, 4, 16), 256, 0, stream>>>(XZ, T1, T2, 256, 256, 256, 15.f, -1.f);
      k_bmm_f32<<<dim3(4, 4, 16), 256, 0, stream>>>(XZ, T2, T1, 256, 256, 256, 13.f, -1.f);
      k_bmm_f32<<<dim3(4, 4, 16), 256, 0, stream>>>(Zc, T1, Zo, 256, 256, 256, 0.f, 0.25f);
      float* tmp = Zc; Zc = Zo; Zo = tmp;
    }
    k_attn3v<<<dim3(256, 16), 256, 0, stream>>>(qkv16, ql16, out3);
    k_bmm_f32<<<dim3(1, 4, 16), 256, 0, stream>>>(Zc, out3, agg, 256, 64, 256, 0.f, 1.f);
    k_attn1_out<<<dim3(128, 16), 256, 0, stream>>>(qkv16, kl16, agg, resw, xout16);
    k_gemm_f16<0, 0><<<dim3(8, 512), 256, 0, stream>>>(xout16, outw, outb, yatt, 32768, 512, 512);
    k_residual<<<dim3((2 * 16130 * 512 + 255) / 256), 256, 0, stream>>>(hbuf, yatt);
  };

  layer(h, l1_g, l1_bt, qkvw16a, outw16a, l1_out_b, l1_res_w);
  k_ppeg<<<dim3(127, 1024), 128, 0, stream>>>(h, h2, w7, b7, w5, b5, w3, b3);
  k_copy_cls<<<dim3(4), 256, 0, stream>>>(h, h2);
  layer(h2, l2_g, l2_bt, qkvw16b, outw16b, l2_out_b, l2_res_w);
  k_final<<<dim3(2), 256, 0, stream>>>(h2, n_g, n_bt, fc2_w, fc2_b, (float*)d_out);
}
